// TransformerMEncoder_27788438405810
// MI455X (gfx1250) — matrix operand order repaired and checked by reading, not compiled, not run on hardware
//
// Transformer-M encoder forward for MI455X (gfx1250), wave32 + WMMA bf16.
// All heavy matmuls go through one batched WMMA GEMM (v_wmma_f32_16x16x32_bf16,
// f32 accumulate). The 262144x128 @ 128x128 Gaussian-RBF GEMM synthesizes its
// A-matrix in the LDS tile loader (AMODE=1) so the 134MB psi tensor is never
// materialized; per-k RBF constants (mean, 1/std, norm) are precomputed in LDS
// so element synthesis is divide-free. Activations are bf16; intermediates
// (~155MB) fit in the 192MB L2. B is staged TRANSPOSED in LDS so both A and B
// fragments load as ds_load_b128.
#include <hip/hip_runtime.h>
#include <hip/hip_bf16.h>
#include <stdint.h>

#define NATOM  512
#define DIM    512
#define NHEAD  16
#define KG     128
#define MEDGE  8
#define NLAYER 4
#define NVALID 513          // n+1 valid rows (cls + atoms)
#define NPAD   544          // 17*32: multiple of 32 (WMMA K) and 16 (M/N)
#define NN     (NATOM*NATOM)
#define HD     (NHEAD*DIM)  // 8192

typedef __attribute__((ext_vector_type(16))) __bf16        v16bf;
typedef __attribute__((ext_vector_type(16))) unsigned short v16u;
typedef __attribute__((ext_vector_type(8)))  float          v8f;

__device__ __forceinline__ unsigned short f2bf(float f) {
  unsigned int u = __float_as_uint(f);
  u += 0x7FFFu + ((u >> 16) & 1u);           // round-to-nearest-even
  return (unsigned short)(u >> 16);
}
__device__ __forceinline__ float bf2f(unsigned short h) {
  return __uint_as_float(((unsigned int)h) << 16);
}

// ---------------- batched WMMA GEMM ----------------
#define BM 128
#define BN 128
#define BK 32
#define TPB 256
#define LDA_S  40   // A tile row stride (halves): 80B, 16B-aligned spans
#define LDBT_S 48   // B^T tile row stride (halves): 96B, 32B-aligned spans, bank-spread

struct GemmP {
  const void* A; const void* B; void* C;
  long long lda, ldb, ldc;
  long long sA, sB, sC;          // batch strides (elements)
  int M, N, K;
  int aBf, bBf, transB;          // transB: B stored [N,K] row-major
  float alpha;
  int epi;                       // 1 = exact GELU
  int storeMode;                 // 0 f32, 1 bf16, 2 f32 "+= transposed" C[gc*ldcT+gr]
  long long ldcT;
  const float* bias; long long sBias; int biasMode; int nValid; int biasN;
  // AMODE==1: synthesize psi(r,k) = normc(k)*exp(-0.5*((xg(r)-mean(k))/std(k))^2)
  const float* dist; const int* et; const float* gtab; const float* btab;
  const float* means; const float* stds;
};

// A 16x32 fragment in this code's contiguous k order (same as frag_b): lanes 0-15 M=lane K{0..15}, lanes 16-31 K{16..31}
__device__ __forceinline__ v16bf frag_a(const unsigned short* sa, int mBase, int lane) {
  const unsigned short* row = sa + (mBase + (lane & 15)) * LDA_S;
  int h16 = ((lane >> 4) & 1) * 16;
  v16u u;
#pragma unroll
  for (int j = 0; j < 16; ++j) {
    u[j] = row[h16 + j];
  }
  return __builtin_bit_cast(v16bf, u);
}
// CDNA5 16-bit B 32x16 fragment from TRANSPOSED LDS tile sBt[n][k]:
// N = lane&15; lanes 0-15 take K=0..15, lanes 16-31 K=16..31 (same k order as frag_a) -> 16 contiguous halves
__device__ __forceinline__ v16bf frag_b(const unsigned short* sbt, int nBase, int lane) {
  const unsigned short* col = sbt + (nBase + (lane & 15)) * LDBT_S + ((lane >> 4) & 1) * 16;
  v16u u;
#pragma unroll
  for (int j = 0; j < 16; ++j) u[j] = col[j];  // 32B contiguous -> 2x ds_load_b128
  return __builtin_bit_cast(v16bf, u);
}

template<int AMODE>
__global__ void __launch_bounds__(TPB) gemm_wmma(GemmP p) {
  __shared__ unsigned short sA [BM * LDA_S];
  __shared__ unsigned short sBt[BN * LDBT_S];
  __shared__ float sxg[BM];
  __shared__ float smean[KG], sisd[KG], snc[KG];   // AMODE==1 per-k RBF constants

  const int tid  = threadIdx.x;
  const int lane = tid & 31;
  const int wave = tid >> 5;
  const int wm = (wave & 3) * 32;   // 4 waves along M
  const int wn = (wave >> 2) * 64;  // 2 waves along N
  const long long rowBase = (long long)blockIdx.y * BM;
  const long long colBase = (long long)blockIdx.x * BN;
  const int bz = blockIdx.z;
  const long long aOff = (long long)bz * p.sA;
  const long long bOff = (long long)bz * p.sB;
  const long long cOff = (long long)bz * p.sC;

  if (AMODE == 1) {
    // per-row xg = gamma[etype]*dist + beta[etype]
    for (int i = tid; i < BM; i += TPB) {
      long long gr = rowBase + i;
      float xg = 0.f;
      if (gr < p.M) {
        int e = p.et[gr];
        xg = p.gtab[e] * p.dist[gr] + p.btab[e];
      }
      sxg[i] = xg;
    }
    // per-k constants: one divide per k per block (instead of 2 per element)
    for (int k = tid; k < p.K; k += TPB) {
      float sd = fabsf(p.stds[k]) + 0.01f;
      float is = 1.0f / sd;
      smean[k] = p.means[k];
      sisd[k]  = is;
      snc[k]   = 0.3989422804014327f * is;
    }
  }

  v8f acc[2][4] = {};
  const int ksteps = p.K / BK;
  for (int kt = 0; kt < ksteps; ++kt) {
    const long long k0 = (long long)kt * BK;
    __syncthreads();
    // ---- stage A tile (f32/bf16 global -> bf16 LDS, or synthesize psi) ----
    for (int i = tid; i < BM * BK; i += TPB) {
      int r = i >> 5, c = i & 31;
      long long gr = rowBase + r;
      long long gk = k0 + c;
      float v = 0.f;
      if (gr < p.M) {
        if (AMODE == 1) {
          float t = (sxg[r] - smean[gk]) * sisd[gk];
          v = snc[gk] * __expf(-0.5f * t * t);
        } else if (p.aBf) {
          v = bf2f(((const unsigned short*)p.A)[aOff + gr * p.lda + gk]);
        } else {
          v = ((const float*)p.A)[aOff + gr * p.lda + gk];
        }
      }
      sA[r * LDA_S + c] = f2bf(v);
    }
    // ---- stage B tile transposed into sBt[n][k] ----
    if (p.transB) {
      // B stored [N,K]: K fastest across lanes -> coalesced global + contiguous LDS
      for (int i = tid; i < BK * BN; i += TPB) {
        int r = i >> 5, c = i & 31;      // r = tile col (n), c = tile k
        long long gc = colBase + r;
        long long gk = k0 + c;
        float v = 0.f;
        if (gc < p.N) {
          long long idx = gc * p.ldb + gk;
          v = p.bBf ? bf2f(((const unsigned short*)p.B)[bOff + idx])
                    : ((const float*)p.B)[bOff + idx];
        }
        sBt[r * LDBT_S + c] = f2bf(v);
      }
      // one global_prefetch_b8 per thread for the next K-step's B rows (L2 -> WGP)
      if (kt + 1 < ksteps && tid < BN) {
        long long gc = colBase + tid;
        if (gc < p.N) {
          long long idx = bOff + gc * p.ldb + (k0 + BK);
          __builtin_prefetch((const char*)p.B + (p.bBf ? 2ll : 4ll) * idx, 0, 1);
        }
      }
    } else {
      // B stored [K,N]: N fastest across lanes -> coalesced global, strided LDS write
      for (int i = tid; i < BK * BN; i += TPB) {
        int kk = i >> 7, c = i & 127;
        long long gk = k0 + kk;
        long long gc = colBase + c;
        float v = 0.f;
        if (gc < p.N) {
          long long idx = gk * p.ldb + gc;
          v = p.bBf ? bf2f(((const unsigned short*)p.B)[bOff + idx])
                    : ((const float*)p.B)[bOff + idx];
        }
        sBt[c * LDBT_S + kk] = f2bf(v);
      }
      if (kt + 1 < ksteps && tid < BK) {
        long long gk = k0 + BK + tid;
        long long idx = bOff + gk * p.ldb + colBase;
        __builtin_prefetch((const char*)p.B + (p.bBf ? 2ll : 4ll) * idx, 0, 1);
      }
    }
    __syncthreads();
    // ---- preload all fragments, then 8 back-to-back WMMAs per wave ----
    v16bf a0 = frag_a(sA, wm,      lane);
    v16bf a1 = frag_a(sA, wm + 16, lane);
    v16bf b0 = frag_b(sBt, wn,      lane);
    v16bf b1 = frag_b(sBt, wn + 16, lane);
    v16bf b2 = frag_b(sBt, wn + 32, lane);
    v16bf b3 = frag_b(sBt, wn + 48, lane);
    acc[0][0] = __builtin_amdgcn_wmma_f32_16x16x32_bf16(false, a0, false, b0, (short)0, acc[0][0], false, false);
    acc[1][0] = __builtin_amdgcn_wmma_f32_16x16x32_bf16(false, a1, false, b0, (short)0, acc[1][0], false, false);
    acc[0][1] = __builtin_amdgcn_wmma_f32_16x16x32_bf16(false, a0, false, b1, (short)0, acc[0][1], false, false);
    acc[1][1] = __builtin_amdgcn_wmma_f32_16x16x32_bf16(false, a1, false, b1, (short)0, acc[1][1], false, false);
    acc[0][2] = __builtin_amdgcn_wmma_f32_16x16x32_bf16(false, a0, false, b2, (short)0, acc[0][2], false, false);
    acc[1][2] = __builtin_amdgcn_wmma_f32_16x16x32_bf16(false, a1, false, b2, (short)0, acc[1][2], false, false);
    acc[0][3] = __builtin_amdgcn_wmma_f32_16x16x32_bf16(false, a0, false, b3, (short)0, acc[0][3], false, false);
    acc[1][3] = __builtin_amdgcn_wmma_f32_16x16x32_bf16(false, a1, false, b3, (short)0, acc[1][3], false, false);
  }

  // ---- epilogue: f32 C 16x16 layout: N=lane&15, M = vgpr + 8*(lane>=16) ----
  const int h8 = ((lane >> 4) & 1) * 8;
  const int nl = lane & 15;
#pragma unroll
  for (int i = 0; i < 2; ++i)
#pragma unroll
    for (int j = 0; j < 4; ++j)
#pragma unroll
      for (int e = 0; e < 8; ++e) {
        long long gr = rowBase + wm + i * 16 + h8 + e;
        long long gc = colBase + wn + j * 16 + nl;
        if (gr >= p.M || gc >= p.N) continue;
        float v = acc[i][j][e] * p.alpha;
        if (p.epi == 1) v = 0.5f * v * (1.0f + erff(v * 0.70710678118f));
        if (p.biasMode) {            // attention: mask padded cols, add graph bias to [1:,1:]
          if (gc >= p.nValid) v = -1e30f;
          else if (gr >= 1 && gr < p.nValid && gc >= 1)
            v += p.bias[(long long)bz * p.sBias + (gr - 1) * (long long)p.biasN + (gc - 1)];
        }
        if (p.storeMode == 0)      ((float*)p.C)[cOff + gr * p.ldc + gc] = v;
        else if (p.storeMode == 1) ((unsigned short*)p.C)[cOff + gr * p.ldc + gc] = f2bf(v);
        else                       ((float*)p.C)[cOff + gc * p.ldcT + gr] += v;
      }
}

// ---------------- small kernels ----------------
__global__ void k_zero_i32(int* p, long long n) {
  long long i = (long long)blockIdx.x * blockDim.x + threadIdx.x;
  if (i < n) p[i] = 0;
}
__global__ void k_scatter(const int* ei, const int* ea, int* et, int E) {
  int e = blockIdx.x * blockDim.x + threadIdx.x;
  if (e >= E) return;
  int f = ei[e], t = ei[E + e], ty = ea[e];
  et[f * NATOM + t] = ty;
  et[t * NATOM + f] = ty;
}
__global__ void k_degree(const int* et, int* deg) {
  __shared__ int red[256];
  int n = blockIdx.x, c = 0;
  for (int m = threadIdx.x; m < NATOM; m += 256) c += (et[n * NATOM + m] != 0);
  red[threadIdx.x] = c; __syncthreads();
  for (int s = 128; s > 0; s >>= 1) { if (threadIdx.x < s) red[threadIdx.x] += red[threadIdx.x + s]; __syncthreads(); }
  if (threadIdx.x == 0) deg[n] = red[0];
}
__global__ void k_dist(const float* pos, float* dist) {
  long long i = (long long)blockIdx.x * blockDim.x + threadIdx.x;
  if (i >= NN) return;
  int n = (int)(i >> 9), m = (int)(i & 511);
  float dd = 1e-12f;
#pragma unroll
  for (int c = 0; c < 3; ++c) { float d = pos[n*3+c] - pos[m*3+c]; dd += d*d; }
  dist[i] = sqrtf(dd);
}
// bias[h,n,m] = spd_tab[sp][h] + (sum_i edge_tab[edge_input[n,m,i]] @ ew[i])[h] / sp'
__global__ void k_bias_init(const int* sp, const int* ein, const float* spdTab,
                            const float* eTab, const float* eDis, float* bias) {
  long long i = (long long)blockIdx.x * blockDim.x + threadIdx.x;
  if (i >= NN) return;
  float acc[NHEAD];
#pragma unroll
  for (int h = 0; h < NHEAD; ++h) acc[h] = 0.f;
  for (int mi = 0; mi < MEDGE; ++mi) {
    int bt = ein[i * MEDGE + mi];
    const float* et = eTab + bt * NHEAD;
    const float* ew = eDis + mi * NHEAD * NHEAD;
    for (int h = 0; h < NHEAD; ++h) {
      float ev = et[h];
#pragma unroll
      for (int k = 0; k < NHEAD; ++k) acc[k] += ev * ew[h * NHEAD + k];
    }
  }
  int spv = sp[i];
  float rspd = (spv == 0) ? 1.f : (1.0f / (float)spv);
#pragma unroll
  for (int h = 0; h < NHEAD; ++h)
    bias[(long long)h * NN + i] = spdTab[spv * NHEAD + h] + acc[h] * rspd;
}
__global__ void k_psisum(const float* dist, const int* et, const float* gtab, const float* btab,
                         const float* means, const float* stds, float* psum) {
  int n = blockIdx.x, k = threadIdx.x;      // blockDim = KG
  float sd  = fabsf(stds[k]) + 0.01f;
  float isd = 1.0f / sd;
  float nc  = 0.3989422804014327f * isd;
  float mean = means[k];
  float s = 0.f;
  for (int m = 0; m < NATOM; ++m) {
    long long i = (long long)n * NATOM + m;
    int e = et[i];
    float xg = gtab[e] * dist[i] + btab[e];
    float t = (xg - mean) * isd;
    s += nc * __expf(-0.5f * t * t);
  }
  psum[(long long)n * KG + k] = s;
}
__global__ void k_xinit(const float* ax, const float* atab, const float* dtab, const int* atoms,
                        const int* deg, const float* p3s, const float* cls,
                        float* x, unsigned short* xbf) {
  int row = blockIdx.x;
  for (int d = threadIdx.x; d < DIM; d += blockDim.x) {
    float v = 0.f;
    if (row == 0) v = cls[d];
    else if (row < NVALID) {
      int a = row - 1;
      v = ax[a*DIM+d] + atab[atoms[a]*DIM+d] + dtab[deg[a]*DIM+d] + p3s[a*DIM+d];
    }
    x[(long long)row * DIM + d] = v;
    xbf[(long long)row * DIM + d] = f2bf(v);
  }
}
__global__ void k_softmax(const float* attn, unsigned short* out) {
  __shared__ float red[256];
  long long base = (long long)blockIdx.x * NPAD;
  float mx = -3e30f;
  for (int c = threadIdx.x; c < NPAD; c += 256) mx = fmaxf(mx, attn[base + c]);
  red[threadIdx.x] = mx; __syncthreads();
  for (int s = 128; s > 0; s >>= 1) { if (threadIdx.x < s) red[threadIdx.x] = fmaxf(red[threadIdx.x], red[threadIdx.x + s]); __syncthreads(); }
  mx = red[0]; __syncthreads();
  float sum = 0.f;
  for (int c = threadIdx.x; c < NPAD; c += 256) sum += __expf(attn[base + c] - mx);
  red[threadIdx.x] = sum; __syncthreads();
  for (int s = 128; s > 0; s >>= 1) { if (threadIdx.x < s) red[threadIdx.x] += red[threadIdx.x + s]; __syncthreads(); }
  float inv = 1.0f / red[0];
  for (int c = threadIdx.x; c < NPAD; c += 256)
    out[base + c] = f2bf(__expf(attn[base + c] - mx) * inv);
}
__global__ void k_ln(float* x, const float* z, const float* s, const float* b, unsigned short* xbf) {
  __shared__ float red[256];
  int row = blockIdx.x, t = threadIdx.x;
  long long base = (long long)row * DIM;
  float y0 = 0.f, y1 = 0.f;
  if (row < NVALID) { y0 = x[base + t] + z[base + t]; y1 = x[base + t + 256] + z[base + t + 256]; }
  red[t] = y0 + y1; __syncthreads();
  for (int ss = 128; ss > 0; ss >>= 1) { if (t < ss) red[t] += red[t + ss]; __syncthreads(); }
  float mu = red[0] * (1.0f / DIM); __syncthreads();
  float d0 = y0 - mu, d1 = y1 - mu;
  red[t] = d0 * d0 + d1 * d1; __syncthreads();
  for (int ss = 128; ss > 0; ss >>= 1) { if (t < ss) red[t] += red[t + ss]; __syncthreads(); }
  float r = rsqrtf(red[0] * (1.0f / DIM) + 1e-5f);
  float o0 = 0.f, o1 = 0.f;
  if (row < NVALID) { o0 = d0 * r * s[t] + b[t]; o1 = d1 * r * s[t + 256] + b[t + 256]; }
  x[base + t] = o0;       xbf[base + t] = f2bf(o0);
  x[base + t + 256] = o1; xbf[base + t + 256] = f2bf(o1);
}
__global__ void k_copy(const float* x, float* out, long long n) {
  long long i = (long long)blockIdx.x * blockDim.x + threadIdx.x;
  if (i < n) out[i] = x[i];
}

// ---------------- host orchestration ----------------
extern "C" void kernel_launch(void* const* d_in, const int* in_sizes, int n_in,
                              void* d_out, int out_size, void* d_ws, size_t ws_size,
                              hipStream_t stream) {
  const float* atoms_x    = (const float*)d_in[0];
  const float* pos        = (const float*)d_in[1];
  const int*   atoms      = (const int*)d_in[2];
  const int*   edge_index = (const int*)d_in[3];
  const int*   edge_attr  = (const int*)d_in[4];
  const int*   spatial    = (const int*)d_in[5];
  const int*   edge_input = (const int*)d_in[6];
  const float* atom_tab   = (const float*)d_in[7];
  const float* degree_tab = (const float*)d_in[8];
  const float* w3d        = (const float*)d_in[9];
  const float* spd_tab    = (const float*)d_in[10];
  const float* edge_tab   = (const float*)d_in[11];
  const float* edge_dis   = (const float*)d_in[12];
  const float* means      = (const float*)d_in[13];
  const float* stds       = (const float*)d_in[14];
  const float* gamma_tab  = (const float*)d_in[15];
  const float* beta_tab   = (const float*)d_in[16];
  const float* pW1        = (const float*)d_in[17];
  const float* pW2        = (const float*)d_in[18];
  const float* cls_token  = (const float*)d_in[19];
  const float* Qw         = (const float*)d_in[20];
  const float* Kw         = (const float*)d_in[21];
  const float* Vw         = (const float*)d_in[22];
  const float* Wout       = (const float*)d_in[23];
  const float* F1         = (const float*)d_in[24];
  const float* F2         = (const float*)d_in[25];
  const float* ln1_s      = (const float*)d_in[26];
  const float* ln1_b      = (const float*)d_in[27];
  const float* ln2_s      = (const float*)d_in[28];
  const float* ln2_b      = (const float*)d_in[29];
  const int E = in_sizes[4];
  (void)n_in; (void)out_size; (void)ws_size;

  char* w = (char*)d_ws;
  size_t off = 0;
  auto alloc = [&](size_t bytes) -> char* {
    char* r = w + off; off += (bytes + 255) & ~(size_t)255; return r;
  };
  int*   etypes = (int*)  alloc((size_t)NN * 4);
  int*   deg    = (int*)  alloc((size_t)NATOM * 4);
  float* dist   = (float*)alloc((size_t)NN * 4);
  float* bias   = (float*)alloc((size_t)NHEAD * NN * 4);
  float* psum   = (float*)alloc((size_t)NATOM * KG * 4);
  float* p3s    = (float*)alloc((size_t)NATOM * DIM * 4);
  unsigned short* g1   = (unsigned short*)alloc((size_t)NN * KG * 2);
  float* x      = (float*)alloc((size_t)NPAD * DIM * 4);
  unsigned short* xbf  = (unsigned short*)alloc((size_t)NPAD * DIM * 2);
  unsigned short* qbf  = (unsigned short*)alloc((size_t)NPAD * HD * 2);
  unsigned short* kbf  = (unsigned short*)alloc((size_t)NPAD * HD * 2);
  unsigned short* vbf  = (unsigned short*)alloc((size_t)NPAD * HD * 2);
  float* attn   = (float*)alloc((size_t)NHEAD * NPAD * NPAD * 4);
  unsigned short* atbf = (unsigned short*)alloc((size_t)NHEAD * NPAD * NPAD * 2);
  unsigned short* obf  = (unsigned short*)alloc((size_t)NPAD * HD * 2);
  float* zbuf   = (float*)alloc((size_t)NPAD * DIM * 4);
  unsigned short* h1bf = (unsigned short*)alloc((size_t)NPAD * DIM * 2);

  auto gemm = [&](const GemmP& p, int batch, int amode) {
    dim3 g((p.N + BN - 1) / BN, (p.M + BM - 1) / BM, batch);
    if (amode) gemm_wmma<1><<<g, TPB, 0, stream>>>(p);
    else       gemm_wmma<0><<<g, TPB, 0, stream>>>(p);
  };

  // graph preprocessing
  k_zero_i32<<<(NN + 255) / 256, 256, 0, stream>>>(etypes, NN);
  k_scatter<<<(E + 255) / 256, 256, 0, stream>>>(edge_index, edge_attr, etypes, E);
  k_degree<<<NATOM, 256, 0, stream>>>(etypes, deg);
  k_dist<<<(NN + 255) / 256, 256, 0, stream>>>(pos, dist);
  k_bias_init<<<(NN + 255) / 256, 256, 0, stream>>>(spatial, edge_input, spd_tab, edge_tab, edge_dis, bias);
  k_psisum<<<NATOM, KG, 0, stream>>>(dist, etypes, gamma_tab, beta_tab, means, stds, psum);

  { // phi_3d_sum = psi_sum @ w3d^T   [512,512]
    GemmP p{}; p.alpha = 1.f;
    p.A = psum; p.lda = KG;
    p.B = w3d;  p.transB = 1; p.ldb = KG;
    p.C = p3s;  p.ldc = DIM;
    p.M = NATOM; p.N = DIM; p.K = KG;
    gemm(p, 1, 0);
  }
  { // g1 = gelu(psi @ pW1^T)  [NN,128] bf16, psi synthesized in loader
    GemmP p{}; p.alpha = 1.f;
    p.B = pW1; p.transB = 1; p.ldb = KG;
    p.C = g1;  p.ldc = KG; p.storeMode = 1; p.epi = 1;
    p.M = NN; p.N = KG; p.K = KG;
    p.dist = dist; p.et = etypes; p.gtab = gamma_tab; p.btab = beta_tab;
    p.means = means; p.stds = stds;
    gemm(p, 1, 1);
  }
  { // bias[h, nm] += (g1 @ pW2^T)[nm, h]   (transposed accumulate)
    GemmP p{}; p.alpha = 1.f;
    p.A = g1;  p.aBf = 1; p.lda = KG;
    p.B = pW2; p.transB = 1; p.ldb = KG;
    p.C = bias; p.storeMode = 2; p.ldcT = NN;
    p.M = NN; p.N = NHEAD; p.K = KG;
    gemm(p, 1, 0);
  }
  k_xinit<<<NPAD, 256, 0, stream>>>(atoms_x, atom_tab, degree_tab, atoms, deg, p3s, cls_token, x, xbf);

  const float scaling = 0.17677669529663687f; // (D/H)^-0.5 = 1/sqrt(32)

  for (int l = 0; l < NLAYER; ++l) {
    const float* Ql  = Qw   + (size_t)l * NHEAD * DIM * DIM;
    const float* Kl  = Kw   + (size_t)l * NHEAD * DIM * DIM;
    const float* Vl  = Vw   + (size_t)l * NHEAD * DIM * DIM;
    const float* Wl  = Wout + (size_t)l * DIM * HD;
    const float* F1l = F1   + (size_t)l * DIM * DIM;
    const float* F2l = F2   + (size_t)l * DIM * DIM;

    for (int which = 0; which < 3; ++which) { // Q/K/V, batched over 16 heads
      GemmP p{}; p.alpha = 1.f;
      p.A = xbf; p.aBf = 1; p.lda = DIM; p.sA = 0;
      p.B = (which == 0 ? Ql : which == 1 ? Kl : Vl);
      p.transB = 1; p.ldb = DIM; p.sB = (long long)DIM * DIM;
      p.C = (which == 0 ? qbf : which == 1 ? kbf : vbf);
      p.ldc = HD; p.sC = DIM; p.storeMode = 1;
      p.M = NPAD; p.N = DIM; p.K = DIM;
      gemm(p, NHEAD, 0);
    }
    { // scores = q @ k^T * scaling (+bias, mask padded cols), batched over heads
      GemmP p{}; p.alpha = scaling;
      p.A = qbf; p.aBf = 1; p.lda = HD; p.sA = DIM;
      p.B = kbf; p.bBf = 1; p.transB = 1; p.ldb = HD; p.sB = DIM;
      p.C = attn; p.ldc = NPAD; p.sC = (long long)NPAD * NPAD;
      p.M = NPAD; p.N = NPAD; p.K = DIM;
      p.bias = bias; p.sBias = NN; p.biasMode = 1; p.nValid = NVALID; p.biasN = NATOM;
      gemm(p, NHEAD, 0);
    }
    k_softmax<<<NHEAD * NPAD, 256, 0, stream>>>(attn, atbf);
    { // o = attn @ v, batched over heads, written interleaved [n, h*D+e]
      GemmP p{}; p.alpha = 1.f;
      p.A = atbf; p.aBf = 1; p.lda = NPAD; p.sA = (long long)NPAD * NPAD;
      p.B = vbf;  p.bBf = 1; p.transB = 0; p.ldb = HD; p.sB = DIM;
      p.C = obf;  p.ldc = HD; p.sC = DIM; p.storeMode = 1;
      p.M = NPAD; p.N = DIM; p.K = NPAD;
      gemm(p, NHEAD, 0);
    }
    { // z = o @ Wout^T   K=8192
      GemmP p{}; p.alpha = 1.f;
      p.A = obf; p.aBf = 1; p.lda = HD;
      p.B = Wl;  p.transB = 1; p.ldb = HD;
      p.C = zbuf; p.ldc = DIM;
      p.M = NPAD; p.N = DIM; p.K = HD;
      gemm(p, 1, 0);
    }
    k_ln<<<NPAD, 256, 0, stream>>>(x, zbuf, ln1_s + l * DIM, ln1_b + l * DIM, xbf);
    { // h1 = gelu(x @ F1^T) bf16
      GemmP p{}; p.alpha = 1.f;
      p.A = xbf; p.aBf = 1; p.lda = DIM;
      p.B = F1l; p.transB = 1; p.ldb = DIM;
      p.C = h1bf; p.ldc = DIM; p.storeMode = 1; p.epi = 1;
      p.M = NPAD; p.N = DIM; p.K = DIM;
      gemm(p, 1, 0);
    }
    { // ff = h1 @ F2^T
      GemmP p{}; p.alpha = 1.f;
      p.A = h1bf; p.aBf = 1; p.lda = DIM;
      p.B = F2l;  p.transB = 1; p.ldb = DIM;
      p.C = zbuf; p.ldc = DIM;
      p.M = NPAD; p.N = DIM; p.K = DIM;
      gemm(p, 1, 0);
    }
    k_ln<<<NPAD, 256, 0, stream>>>(x, zbuf, ln2_s + l * DIM, ln2_b + l * DIM, xbf);
  }
  k_copy<<<(int)(((long long)NVALID * DIM + 255) / 256), 256, 0, stream>>>(
      x, (float*)d_out, (long long)NVALID * DIM);
}
